// AudioGPT_77902116815076
// MI455X (gfx1250) — compile-verified, matrix operand order checked by reading, not run on hardware
//
#include <hip/hip_runtime.h>
#include <hip/hip_bf16.h>
#include <math.h>

// ---------------------------------------------------------------------------
// AudioGPT forward on MI455X (gfx1250, wave32, WMMA bf16, async LDS copies,
// double-buffered pipelines).
// B=2 T=2048 V=256 C=1024 H=16 DH=64 L=8
//
// - one-time fp32->bf16 weight convert+transpose into d_ws
// - bf16 activations end-to-end, fp32 WMMA accumulation
// - all LDS staging via GLOBAL_LOAD_ASYNC_TO_LDS_B128 (ASYNCcnt-tracked,
//   in-order completion) with double buffering: tile i+1 copies in flight
//   while tile i feeds v_wmma_f32_16x16x32_bf16.
// ---------------------------------------------------------------------------

typedef __attribute__((ext_vector_type(16))) __bf16 v16bf;
typedef __attribute__((ext_vector_type(8)))  __bf16 v8bf;
typedef __attribute__((ext_vector_type(4)))  __bf16 v4bf;
typedef __attribute__((ext_vector_type(8)))  float  v8f;

#define WMMA_BF16(a, b, c) \
    __builtin_amdgcn_wmma_f32_16x16x32_bf16(false, (a), false, (b), (short)0, (c), false, false)

__device__ __forceinline__ __bf16 f2bf(float f) { return (__bf16)f; }

// LDS byte offset of a __shared__ pointer (flat LDS aperture: low 32 bits).
__device__ __forceinline__ unsigned lds_u32(const void* p) {
    return (unsigned)(unsigned long long)p;
}

// Async copy of 16 bytes (per lane) global -> LDS, tracked by ASYNCcnt.
__device__ __forceinline__ void async_b128(unsigned lds_addr, const void* base,
                                           unsigned goff) {
    asm volatile("global_load_async_to_lds_b128 %0, %1, %2 offset:0"
                 :: "v"(lds_addr), "v"(goff), "s"(base) : "memory");
}
// Wait until this wave's outstanding async copies <= N (in-order completion).
__device__ __forceinline__ void wait_async0() {
    asm volatile("s_wait_asynccnt 0" ::: "memory");
}
__device__ __forceinline__ void wait_async4() {
    asm volatile("s_wait_asynccnt 4" ::: "memory");
}

// Load a 16x32 bf16 A/B fragment for this lane from LDS (row-major tile).
// Elements 0..7  <-> K = kofs + 0..7   (one ds_load_b128)
// Elements 8..15 <-> K = kofs + 16..23 (one ds_load_b128)
__device__ __forceinline__ v16bf frag_ld(const __bf16* rowbase, int kofs) {
    union { v16bf v; v8bf h[2]; } u;
    u.h[0] = *(const v8bf*)(rowbase + kofs);
    u.h[1] = *(const v8bf*)(rowbase + kofs + 16);
    return u.v;
}

// ---------------------------------------------------------------------------
// One-time weight prep: dst[c][r] = (bf16)src[r][c]  (per layer z)
// ---------------------------------------------------------------------------
__global__ __launch_bounds__(256) void transpose_cast_kernel(
    const float* __restrict__ src, __bf16* __restrict__ dst,
    int R, int Ccols, size_t srcStride, size_t dstStride)
{
    __shared__ float tile[32][33];
    const float* s = src + blockIdx.z * srcStride;
    __bf16* d = dst + blockIdx.z * dstStride;
    const int r0 = blockIdx.y * 32, c0 = blockIdx.x * 32;
    const int tx = threadIdx.x & 31, tg = threadIdx.x >> 5;
#pragma unroll
    for (int i = 0; i < 4; ++i) {
        int r = tg + i * 8;
        tile[r][tx] = s[(size_t)(r0 + r) * Ccols + c0 + tx];
    }
    __syncthreads();
#pragma unroll
    for (int i = 0; i < 4; ++i) {
        int c = tg + i * 8;
        d[(size_t)(c0 + c) * R + r0 + tx] = f2bf(tile[tx][c]);
    }
}

__global__ __launch_bounds__(256) void cast_f32_bf16_kernel(
    const float* __restrict__ src, __bf16* __restrict__ dst)
{
    int i = (blockIdx.x * 256 + threadIdx.x) * 4;
    float4 f = *(const float4*)(src + i);
    v4bf h = {f2bf(f.x), f2bf(f.y), f2bf(f.z), f2bf(f.w)};
    *(v4bf*)(dst + i) = h;
}

// ---------------------------------------------------------------------------
// 128x128-tile bf16 WMMA GEMM: A[M][K] bf16 row-major, Bt[N][K] bf16
// (pre-transposed), + fp32 bias[N].  Double-buffered async LDS staging.
// MODE 0: C (bf16)        MODE 1: embed (+pos, bf16)
// MODE 2: qkv scatter (q,k: [B,H,T,DH]; v transposed: [B,H,DH,T]) bf16
// MODE 3: Cf (fp32, final logits)
// 256 threads = 8 waves; wave (wm,wn) owns 64x32 => 4x2 accumulators.
// ---------------------------------------------------------------------------
template <int MODE>
__global__ __launch_bounds__(256) void gemm_bf16_wmma(
    const __bf16* __restrict__ A, const __bf16* __restrict__ Bt,
    const float* __restrict__ bias, const float* __restrict__ pos,
    __bf16* __restrict__ C, float* __restrict__ Cf,
    __bf16* __restrict__ Qo, __bf16* __restrict__ Ko, __bf16* __restrict__ Vo,
    int M, int N, int K)
{
    __shared__ __bf16 lA[2][128 * 40];   // [row][k], stride 40 (pad)
    __shared__ __bf16 lB[2][128 * 40];   // [n][k]

    const int tid  = threadIdx.x;
    const int lane = tid & 31;
    const int wave = tid >> 5;
    const int half = lane >> 4;
    const int l16  = lane & 15;
    const int wm   = wave >> 2;
    const int wn   = wave & 3;
    const int bm   = blockIdx.y * 128;
    const int bn   = blockIdx.x * 128;

    // Per-thread staging coordinates: 2 x b128 for A, 2 x b128 for B.
    const int ia0 = tid * 8, ia1 = 2048 + tid * 8;
    const int r0 = ia0 >> 5, c0 = ia0 & 31;
    const int r1 = ia1 >> 5, c1 = ia1 & 31;
    const unsigned off0 = (unsigned)(r0 * 40 + c0) * 2;
    const unsigned off1 = (unsigned)(r1 * 40 + c1) * 2;
    const unsigned aBase[2] = {lds_u32(&lA[0][0]), lds_u32(&lA[1][0])};
    const unsigned bBase[2] = {lds_u32(&lB[0][0]), lds_u32(&lB[1][0])};
    const unsigned aG0 = (unsigned)(((size_t)(bm + r0) * K + c0) * 2);
    const unsigned aG1 = (unsigned)(((size_t)(bm + r1) * K + c1) * 2);
    const unsigned bG0 = (unsigned)(((size_t)(bn + r0) * K + c0) * 2);
    const unsigned bG1 = (unsigned)(((size_t)(bn + r1) * K + c1) * 2);

#define GEMM_ISSUE(buf, k0)                                   \
    do {                                                      \
        unsigned kb = (unsigned)(k0) * 2;                     \
        async_b128(aBase[buf] + off0, A, aG0 + kb);           \
        async_b128(aBase[buf] + off1, A, aG1 + kb);           \
        async_b128(bBase[buf] + off0, Bt, bG0 + kb);          \
        async_b128(bBase[buf] + off1, Bt, bG1 + kb);          \
    } while (0)

    v8f acc[4][2];
    const v8f vzero = {0.f, 0.f, 0.f, 0.f, 0.f, 0.f, 0.f, 0.f};
#pragma unroll
    for (int i = 0; i < 4; ++i)
#pragma unroll
        for (int j = 0; j < 2; ++j) acc[i][j] = vzero;

    const int ntiles = K >> 5;
    GEMM_ISSUE(0, 0);
    for (int it = 0; it < ntiles; ++it) {
        const int cur = it & 1;
        if (it + 1 < ntiles) {
            GEMM_ISSUE(1 - cur, (it + 1) * 32);
            wait_async4();     // tile `it` landed; tile `it+1` in flight
        } else {
            wait_async0();
        }
        __syncthreads();

        const __bf16* cA = lA[cur];
        const __bf16* cB = lB[cur];
        v16bf af[4], bf[2];
#pragma unroll
        for (int i = 0; i < 4; ++i)
            af[i] = frag_ld(cA + (wm * 64 + i * 16 + l16) * 40, half * 8);
#pragma unroll
        for (int j = 0; j < 2; ++j)
            bf[j] = frag_ld(cB + (wn * 32 + j * 16 + l16) * 40, half * 8);
#pragma unroll
        for (int i = 0; i < 4; ++i)
#pragma unroll
            for (int j = 0; j < 2; ++j)
                acc[i][j] = WMMA_BF16(af[i], bf[j], acc[i][j]);

        __syncthreads();       // all waves done with `cur` before it refills
    }
#undef GEMM_ISSUE

    // Epilogue. D layout: lane l16 = n%16, VGPR r + 8*half = m%16.
#pragma unroll
    for (int i = 0; i < 4; ++i) {
#pragma unroll
        for (int j = 0; j < 2; ++j) {
#pragma unroll
            for (int r = 0; r < 8; ++r) {
                int m = bm + wm * 64 + i * 16 + r + 8 * half;
                int n = bn + wn * 32 + j * 16 + l16;
                float val = acc[i][j][r] + bias[n];
                if (MODE == 0) {
                    C[(size_t)m * N + n] = f2bf(val);
                } else if (MODE == 1) {
                    int t = m & 2047;
                    C[(size_t)m * N + n] = f2bf(val + pos[(size_t)t * 1024 + n]);
                } else if (MODE == 2) {
                    int sel = n >> 10, c = n & 1023;
                    int hh = c >> 6, d = c & 63;
                    int b = m >> 11, t = m & 2047;
                    if (sel == 0)
                        Qo[(((size_t)(b * 16 + hh) * 2048 + t) * 64) + d] = f2bf(val);
                    else if (sel == 1)
                        Ko[(((size_t)(b * 16 + hh) * 2048 + t) * 64) + d] = f2bf(val);
                    else  // V transposed: [B,H,DH,T]
                        Vo[(((size_t)(b * 16 + hh) * 64 + d) * 2048) + t] = f2bf(val);
                } else {
                    Cf[(size_t)m * N + n] = val;
                }
            }
        }
    }
}

// ---------------------------------------------------------------------------
// Flash attention (non-causal). Q,K bf16 [B,H,T,DH]; V bf16 [B,H,DH,T].
// Y out bf16 [B,T,C]. grid = (T/128, B*H); 8 waves x 16 query rows.
// K chunk staged [key][d], V chunk staged [d][key]: linear async copies,
// double-buffered. 1/8 softmax scale pre-folded into Q fragments (exact in
// bf16: power-of-two scale).
// ---------------------------------------------------------------------------
__global__ __launch_bounds__(256) void attn_flash_wmma(
    const __bf16* __restrict__ Qp, const __bf16* __restrict__ Kp,
    const __bf16* __restrict__ Vp, __bf16* __restrict__ Y)
{
    __shared__ __bf16 lK[2][64 * 72];     // [key][d]
    __shared__ __bf16 lV[2][64 * 72];     // [d][key]
    __shared__ __bf16 lP[8 * 16 * 72];    // per-wave [qrow][key]

    const int tid  = threadIdx.x;
    const int lane = tid & 31;
    const int wave = tid >> 5;
    const int half = lane >> 4;
    const int l16  = lane & 15;
    const int bh   = blockIdx.y;
    const int b    = bh >> 4;
    const int h    = bh & 15;
    const int qrow0 = blockIdx.x * 128 + wave * 16;

    const __bf16* qbh = Qp + (size_t)bh * 2048 * 64;
    const __bf16* kbh = Kp + (size_t)bh * 2048 * 64;
    const __bf16* vbh = Vp + (size_t)bh * 2048 * 64;  // [d][t]

    // Preload Q fragments (row = qrow0 + l16), prescale by 1/8 (exact).
    const __bf16* qrowp = qbh + (size_t)(qrow0 + l16) * 64;
    v16bf qf[2];
#pragma unroll
    for (int ks = 0; ks < 2; ++ks) {
        union { v16bf v; v8bf hh[2]; __bf16 e[16]; } u;
        u.hh[0] = *(const v8bf*)(qrowp + ks * 32 + half * 8);
        u.hh[1] = *(const v8bf*)(qrowp + ks * 32 + half * 8 + 16);
#pragma unroll
        for (int e = 0; e < 16; ++e) u.e[e] = f2bf((float)u.e[e] * 0.125f);
        qf[ks] = u.v;
    }

    // Staging coordinates (64x64 bf16 per matrix; 2 x b128 per thread each).
    const int i0 = tid * 8, i1 = 2048 + tid * 8;
    const int kr0 = i0 >> 6, kc0c = i0 & 63;
    const int kr1 = i1 >> 6, kc1c = i1 & 63;
    const unsigned off0 = (unsigned)(kr0 * 72 + kc0c) * 2;
    const unsigned off1 = (unsigned)(kr1 * 72 + kc1c) * 2;
    const unsigned kBase[2] = {lds_u32(&lK[0][0]), lds_u32(&lK[1][0])};
    const unsigned vBase[2] = {lds_u32(&lV[0][0]), lds_u32(&lV[1][0])};
    const unsigned kG0 = (unsigned)(((size_t)kr0 * 64 + kc0c) * 2);     // +kc*128
    const unsigned kG1 = (unsigned)(((size_t)kr1 * 64 + kc1c) * 2);
    const unsigned vG0 = (unsigned)(((size_t)kr0 * 2048 + kc0c) * 2);   // +kc*2
    const unsigned vG1 = (unsigned)(((size_t)kr1 * 2048 + kc1c) * 2);

#define ATTN_ISSUE(buf, kc)                                            \
    do {                                                               \
        async_b128(kBase[buf] + off0, kbh, kG0 + (unsigned)(kc) * 128);\
        async_b128(kBase[buf] + off1, kbh, kG1 + (unsigned)(kc) * 128);\
        async_b128(vBase[buf] + off0, vbh, vG0 + (unsigned)(kc) * 2);  \
        async_b128(vBase[buf] + off1, vbh, vG1 + (unsigned)(kc) * 2);  \
    } while (0)

    float rmax[8], rsum[8];
#pragma unroll
    for (int r = 0; r < 8; ++r) { rmax[r] = -1e30f; rsum[r] = 0.f; }
    const v8f vzero = {0.f, 0.f, 0.f, 0.f, 0.f, 0.f, 0.f, 0.f};
    v8f yacc[4];
#pragma unroll
    for (int nt = 0; nt < 4; ++nt) yacc[nt] = vzero;

    ATTN_ISSUE(0, 0);
    for (int it = 0; it < 32; ++it) {
        const int cur = it & 1;
        if (it + 1 < 32) {
            ATTN_ISSUE(1 - cur, (it + 1) * 64);
            wait_async4();
        } else {
            wait_async0();
        }
        __syncthreads();

        const __bf16* cK = lK[cur];
        const __bf16* cV = lV[cur];

        // S = (Q/8) K^T (fp32 accum).
        v8f sacc[4];
#pragma unroll
        for (int nt = 0; nt < 4; ++nt) sacc[nt] = vzero;
#pragma unroll
        for (int ks = 0; ks < 2; ++ks) {
#pragma unroll
            for (int nt = 0; nt < 4; ++nt) {
                v16bf kf = frag_ld(cK + (nt * 16 + l16) * 72, ks * 32 + half * 8);
                sacc[nt] = WMMA_BF16(qf[ks], kf, sacc[nt]);
            }
        }

        // Online softmax (rows span 16 lanes; wave32 xor reductions).
        __bf16* lPw = lP + wave * 16 * 72;
#pragma unroll
        for (int r = 0; r < 8; ++r) {
            float mx = fmaxf(fmaxf(sacc[0][r], sacc[1][r]),
                             fmaxf(sacc[2][r], sacc[3][r]));
            mx = fmaxf(mx, __shfl_xor(mx, 1));
            mx = fmaxf(mx, __shfl_xor(mx, 2));
            mx = fmaxf(mx, __shfl_xor(mx, 4));
            mx = fmaxf(mx, __shfl_xor(mx, 8));
            float nmax = fmaxf(rmax[r], mx);
            float psum = 0.f;
#pragma unroll
            for (int nt = 0; nt < 4; ++nt) {
                float p = __expf(sacc[nt][r] - nmax);
                psum += p;
                lPw[(r + 8 * half) * 72 + nt * 16 + l16] = f2bf(p);
            }
            psum += __shfl_xor(psum, 1);
            psum += __shfl_xor(psum, 2);
            psum += __shfl_xor(psum, 4);
            psum += __shfl_xor(psum, 8);
            float fac = __expf(rmax[r] - nmax);
            rsum[r] = rsum[r] * fac + psum;
            rmax[r] = nmax;
#pragma unroll
            for (int nt = 0; nt < 4; ++nt) yacc[nt][r] *= fac;
        }

        // Y += P * V.
#pragma unroll
        for (int ks = 0; ks < 2; ++ks) {
            v16bf pf = frag_ld(lPw + l16 * 72, ks * 32 + half * 8);
#pragma unroll
            for (int nt = 0; nt < 4; ++nt) {
                v16bf vf = frag_ld(cV + (nt * 16 + l16) * 72, ks * 32 + half * 8);
                yacc[nt] = WMMA_BF16(pf, vf, yacc[nt]);
            }
        }
        __syncthreads();   // all waves done with `cur` before it refills
    }
#undef ATTN_ISSUE

    // Epilogue: y /= rowsum ; bf16 out to [B,T,C], c = h*64 + d.
#pragma unroll
    for (int nt = 0; nt < 4; ++nt) {
#pragma unroll
        for (int r = 0; r < 8; ++r) {
            int row = qrow0 + r + 8 * half;
            int d   = nt * 16 + l16;
            Y[((size_t)(b * 2048 + row) * 1024) + h * 64 + d] =
                f2bf(yacc[nt][r] / rsum[r]);
        }
    }
}

// ---------------------------------------------------------------------------
// LayerNorm over last dim (1024), bf16 in/out, fp32 math. 1 block per row.
// ---------------------------------------------------------------------------
__global__ __launch_bounds__(256) void layernorm_kernel(
    const __bf16* __restrict__ in, const float* __restrict__ gamma,
    const float* __restrict__ beta, __bf16* __restrict__ out)
{
    __shared__ float red[256];
    const int tid = threadIdx.x;
    const size_t row = blockIdx.x;
    v4bf hv = *(const v4bf*)(in + row * 1024 + tid * 4);
    float x0 = (float)hv[0], x1 = (float)hv[1], x2 = (float)hv[2], x3 = (float)hv[3];

    red[tid] = x0 + x1 + x2 + x3;
    __syncthreads();
    for (int off = 128; off > 0; off >>= 1) {
        if (tid < off) red[tid] += red[tid + off];
        __syncthreads();
    }
    float mu = red[0] * (1.0f / 1024.0f);
    __syncthreads();

    float d0 = x0 - mu, d1 = x1 - mu, d2 = x2 - mu, d3 = x3 - mu;
    red[tid] = d0 * d0 + d1 * d1 + d2 * d2 + d3 * d3;
    __syncthreads();
    for (int off = 128; off > 0; off >>= 1) {
        if (tid < off) red[tid] += red[tid + off];
        __syncthreads();
    }
    float inv = rsqrtf(red[0] * (1.0f / 1024.0f) + 1e-5f);

    int c = tid * 4;
    __bf16* o = out + row * 1024 + c;
    v4bf ho = {f2bf(d0 * inv * gamma[c + 0] + beta[c + 0]),
               f2bf(d1 * inv * gamma[c + 1] + beta[c + 1]),
               f2bf(d2 * inv * gamma[c + 2] + beta[c + 2]),
               f2bf(d3 * inv * gamma[c + 3] + beta[c + 3])};
    *(v4bf*)o = ho;
}

// ---------------------------------------------------------------------------
// Host orchestration (graph-capture safe: only kernel launches on `stream`).
// ---------------------------------------------------------------------------
extern "C" void kernel_launch(void* const* d_in, const int* in_sizes, int n_in,
                              void* d_out, int out_size, void* d_ws, size_t ws_size,
                              hipStream_t stream) {
    const float* x     = (const float*)d_in[0];
    const float* W_emb = (const float*)d_in[1];
    const float* b_emb = (const float*)d_in[2];
    const float* pos   = (const float*)d_in[3];
    const float* Wqkv  = (const float*)d_in[4];
    const float* bqkv  = (const float*)d_in[5];
    const float* Wproj = (const float*)d_in[6];
    const float* bproj = (const float*)d_in[7];
    const float* gamma = (const float*)d_in[8];
    const float* beta  = (const float*)d_in[9];
    const float* W_out = (const float*)d_in[10];
    const float* b_out = (const float*)d_in[11];
    float* out = (float*)d_out;

    // Bump allocator over d_ws (256-byte aligned regions).
    char* w = (char*)d_ws;
    auto alloc = [&](size_t elems) {
        char* p = w;
        w += (elems * sizeof(__bf16) + 255) & ~(size_t)255;
        return (__bf16*)p;
    };
    __bf16* bx    = alloc((size_t)4096 * 256);          // x cast
    __bf16* bWe   = alloc((size_t)1024 * 256);          // W_emb^T
    __bf16* bWqkv = alloc((size_t)8 * 3072 * 1024);     // Wqkv^T per layer
    __bf16* bWp   = alloc((size_t)8 * 1024 * 1024);     // Wproj^T per layer
    __bf16* bWo   = alloc((size_t)256 * 1024);          // W_out^T
    __bf16* h0    = alloc((size_t)4096 * 1024);
    __bf16* h1    = alloc((size_t)4096 * 1024);
    __bf16* q     = alloc((size_t)4096 * 1024);
    __bf16* k     = alloc((size_t)4096 * 1024);
    __bf16* v     = alloc((size_t)4096 * 1024);
    __bf16* y     = alloc((size_t)4096 * 1024);

    dim3 blk(256);

    // ---- One-time (per call) weight prep ----
    cast_f32_bf16_kernel<<<dim3(1024), blk, 0, stream>>>(x, bx);
    transpose_cast_kernel<<<dim3(32, 8, 1), blk, 0, stream>>>(
        W_emb, bWe, 256, 1024, 0, 0);
    transpose_cast_kernel<<<dim3(96, 32, 8), blk, 0, stream>>>(
        Wqkv, bWqkv, 1024, 3072, (size_t)1024 * 3072, (size_t)3072 * 1024);
    transpose_cast_kernel<<<dim3(32, 32, 8), blk, 0, stream>>>(
        Wproj, bWp, 1024, 1024, (size_t)1024 * 1024, (size_t)1024 * 1024);
    transpose_cast_kernel<<<dim3(8, 32, 1), blk, 0, stream>>>(
        W_out, bWo, 1024, 256, 0, 0);

    // ---- Embedding: h0 = x @ W_emb + b_emb + pos  (M=4096 N=1024 K=256) ----
    gemm_bf16_wmma<1><<<dim3(8, 32), blk, 0, stream>>>(
        bx, bWe, b_emb, pos, h0, nullptr, nullptr, nullptr, nullptr,
        4096, 1024, 256);

    __bf16* hin = h0;
    __bf16* hout = h1;
    for (int l = 0; l < 8; ++l) {
        gemm_bf16_wmma<2><<<dim3(24, 32), blk, 0, stream>>>(
            hin, bWqkv + (size_t)l * 3072 * 1024, bqkv + (size_t)l * 3072,
            nullptr, nullptr, nullptr, q, k, v, 4096, 3072, 1024);
        attn_flash_wmma<<<dim3(16, 32), blk, 0, stream>>>(q, k, v, y);
        gemm_bf16_wmma<0><<<dim3(8, 32), blk, 0, stream>>>(
            y, bWp + (size_t)l * 1024 * 1024, bproj + (size_t)l * 1024,
            nullptr, hout, nullptr, nullptr, nullptr, nullptr, 4096, 1024, 1024);
        __bf16* t = hin; hin = hout; hout = t;
    }

    // ---- LayerNorm (hin -> y reused) ----
    layernorm_kernel<<<dim3(4096), blk, 0, stream>>>(hin, gamma, beta, y);

    // ---- Output projection: out = hn @ W_out + b_out (M=4096 N=256 K=1024) ----
    gemm_bf16_wmma<3><<<dim3(2, 32), blk, 0, stream>>>(
        y, bWo, b_out, nullptr, nullptr, out, nullptr, nullptr, nullptr,
        4096, 256, 1024);
}
